// ParametricGTCNN_12524124635993
// MI455X (gfx1250) — matrix operand order repaired and checked by reading, not compiled, not run on hardware
//
#include <hip/hip_runtime.h>
#include <hip/hip_bf16.h>

// Problem constants (match reference)
constexpr int kN   = 2000;
constexpr int kT   = 12;
constexpr int kB   = 4;
constexpr int kNT  = kN * kT;      // 24000
constexpr int kHID = 64;
constexpr int kD2  = 192;          // (K+1)*HID

typedef __attribute__((ext_vector_type(16))) _Float16 v16h;
typedef __attribute__((ext_vector_type(8)))  _Float16 v8h;
typedef __attribute__((ext_vector_type(8)))  float    v8f;

// ---------------- degree + normalized edge values ----------------
__global__ void deg_accum(float* __restrict__ deg, const int* __restrict__ r,
                          int ne, const float* __restrict__ s) {
    int e = blockIdx.x * blockDim.x + threadIdx.x;
    if (e < ne) {
        float v = fmaxf(s[0], 0.0f);
        atomicAdd(&deg[r[e]], v);
    }
}

__global__ void edge_norm(float* __restrict__ vn, const float* __restrict__ deg,
                          const int* __restrict__ r, const int* __restrict__ c,
                          int ne, const float* __restrict__ s) {
    int e = blockIdx.x * blockDim.x + threadIdx.x;
    if (e < ne) {
        float v  = fmaxf(s[0], 0.0f);
        float dr = fmaxf(deg[r[e]], 1e-12f);
        float dc = fmaxf(deg[c[e]], 1e-12f);
        vn[e] = v * rsqrtf(dr * dc);
    }
}

// ---------------- input transpose: x[b,0,n,t] -> Xp[t*N+n] ----------------
__global__ void transpose_x(float* __restrict__ Xp, const float* __restrict__ x, int b) {
    int i = blockIdx.x * blockDim.x + threadIdx.x;   // i = t*N + n
    if (i < kNT) {
        int t = i / kN, n = i - t * kN;
        Xp[i] = x[(size_t)b * kN * kT + (size_t)n * kT + t];
    }
}

// ---------------- SpMM, F = 1 ----------------
__global__ void spmm_f1(float* __restrict__ Y, const float* __restrict__ X,
                        const float* __restrict__ vn, const int* __restrict__ r,
                        const int* __restrict__ c, int ne) {
    int e = blockIdx.x * blockDim.x + threadIdx.x;
    if (e < ne) atomicAdd(&Y[r[e]], vn[e] * X[c[e]]);
}

// ---------------- SpMM, F = 64 ----------------
__global__ void spmm_f64(float* __restrict__ Y, const float* __restrict__ X,
                         const float* __restrict__ vn, const int* __restrict__ r,
                         const int* __restrict__ c, int ne) {
    long long idx = (long long)blockIdx.x * blockDim.x + threadIdx.x;
    int e = (int)(idx >> 6);
    int h = (int)(idx & 63);
    if (e < ne) atomicAdd(&Y[(size_t)r[e] * kHID + h], vn[e] * X[(size_t)c[e] * kHID + h]);
}

// ---------------- layer 1: H1 = relu([X, AX, A2X] @ W1 + b1), d_in = 3 ----------------
__global__ void layer1(float* __restrict__ H1, const float* __restrict__ Xp,
                       const float* __restrict__ Z1, const float* __restrict__ Z2,
                       const float* __restrict__ W1, const float* __restrict__ b1) {
    int idx = blockIdx.x * blockDim.x + threadIdx.x;
    if (idx < kNT * kHID) {
        int i = idx >> 6, h = idx & 63;
        float z = Xp[i] * W1[h] + Z1[i] * W1[kHID + h] + Z2[i] * W1[2 * kHID + h] + b1[h];
        H1[idx] = fmaxf(z, 0.0f);
    }
}

// ---------------- f16 concat buffer: Zc[i, 0:64|64:128|128:192] = H1|S1|S2 ----------------
__global__ void cvt_cat(_Float16* __restrict__ Zc, const float* __restrict__ H1,
                        const float* __restrict__ S1, const float* __restrict__ S2) {
    int idx = blockIdx.x * blockDim.x + threadIdx.x;
    if (idx < kNT * kHID) {
        int i = idx >> 6, h = idx & 63;
        size_t base = (size_t)i * kD2;
        Zc[base + h]            = (_Float16)H1[idx];
        Zc[base + kHID + h]     = (_Float16)S1[idx];
        Zc[base + 2 * kHID + h] = (_Float16)S2[idx];
    }
}

// ---------------- W2 [192,64] -> W2T f16 [64,192] ----------------
__global__ void cvt_w2t(_Float16* __restrict__ W2T, const float* __restrict__ W2) {
    int idx = blockIdx.x * blockDim.x + threadIdx.x;   // 192*64
    if (idx < kD2 * kHID) {
        int k = idx / kHID, h = idx - k * kHID;
        W2T[(size_t)h * kD2 + k] = (_Float16)W2[idx];
    }
}

// ---------------- layer-2 GEMM: H2 = relu(Zc[NT,192] @ W2[192,64] + b2) via WMMA ----------------
// Each wave computes a 16x64 output tile: 4 accumulators, 6 k-steps of 32.
__global__ __launch_bounds__(256)
void gemm_l2_wmma(float* __restrict__ H2, const _Float16* __restrict__ A,
                  const _Float16* __restrict__ Bt, const float* __restrict__ b2) {
    const int wave = threadIdx.x >> 5;
    const int lane = threadIdx.x & 31;
    const int tile = blockIdx.x * 8 + wave;          // M-tile
    if (tile >= kNT / 16) return;                    // wave-uniform exit (EXEC all-1s preserved)
    const int m0   = tile * 16;
    const int half = lane >> 4;                      // 0 | 1
    const int l    = lane & 15;

    v8f acc[4] = {v8f{}, v8f{}, v8f{}, v8f{}};

    const _Float16* arow = A + (size_t)(m0 + l) * kD2;

    #pragma unroll
    for (int ks = 0; ks < 6; ++ks) {
        // A fragment 16x32 f16: lane<16 holds K = k0+[0..7], k0+[16..23]; lane>=16 shifts by 8
        const int ka = ks * 32 + half * 8;
        v8h alo = *(const v8h*)(arow + ka);
        v8h ahi = *(const v8h*)(arow + ka + 16);
        v16h av = __builtin_shufflevector(alo, ahi,
                    0, 1, 2, 3, 4, 5, 6, 7, 8, 9, 10, 11, 12, 13, 14, 15);

        // B fragment 32x16 f16: lane<16 -> col=l, K = k0+[0..7], k0+[16..23]; lane>=16 shifts by 8
        const int kb = ks * 32 + half * 8;
        #pragma unroll
        for (int nb = 0; nb < 4; ++nb) {
            const _Float16* bp = Bt + (size_t)(nb * 16 + l) * kD2 + kb;
            v8h blo = *(const v8h*)(bp);
            v8h bhi = *(const v8h*)(bp + 16);
            v16h bv = __builtin_shufflevector(blo, bhi,
                        0, 1, 2, 3, 4, 5, 6, 7, 8, 9, 10, 11, 12, 13, 14, 15);
            acc[nb] = __builtin_amdgcn_wmma_f32_16x16x32_f16(
                          false, av, false, bv, (short)0, acc[nb], false, false);
        }
    }

    // D layout: VGPR v, lane (half,l): row = m0 + v + half*8, col = nb*16 + l
    #pragma unroll
    for (int nb = 0; nb < 4; ++nb) {
        const int col  = nb * 16 + l;
        const float bb = b2[col];
        #pragma unroll
        for (int v = 0; v < 8; ++v) {
            const int row = m0 + v + half * 8;
            float z = acc[nb][v] + bb;
            H2[(size_t)row * kHID + col] = fmaxf(z, 0.0f);
        }
    }
}

// ---------------- mean-pool over time + head ----------------
__global__ void head_pool(float* __restrict__ out, const float* __restrict__ H2,
                          const float* __restrict__ Wh, const float* __restrict__ bh) {
    int n = blockIdx.x * blockDim.x + threadIdx.x;
    if (n < kN) {
        float acc = 0.0f;
        for (int t = 0; t < kT; ++t) {
            const float* row = H2 + ((size_t)t * kN + n) * kHID;
            #pragma unroll
            for (int h = 0; h < kHID; ++h) acc += row[h] * Wh[h];
        }
        out[n] = acc * (1.0f / kT) + bh[0];
    }
}

extern "C" void kernel_launch(void* const* d_in, const int* in_sizes, int n_in,
                              void* d_out, int out_size, void* d_ws, size_t ws_size,
                              hipStream_t stream) {
    const float* x  = (const float*)d_in[0];
    const float* W1 = (const float*)d_in[1];
    const float* b1 = (const float*)d_in[2];
    const float* W2 = (const float*)d_in[3];
    const float* b2 = (const float*)d_in[4];
    const float* Wh = (const float*)d_in[5];
    const float* bh = (const float*)d_in[6];
    const float* sg[4] = {(const float*)d_in[7],  (const float*)d_in[8],
                          (const float*)d_in[9],  (const float*)d_in[10]};
    const int*   rg[4] = {(const int*)d_in[11], (const int*)d_in[13],
                          (const int*)d_in[15], (const int*)d_in[17]};
    const int*   cg[4] = {(const int*)d_in[12], (const int*)d_in[14],
                          (const int*)d_in[16], (const int*)d_in[18]};
    int ne[4], off[4], E = 0;
    for (int g = 0; g < 4; ++g) { ne[g] = in_sizes[11 + 2 * g]; off[g] = E; E += ne[g]; }

    // ---- workspace carve-out (256B aligned) ----
    char* w = (char*)d_ws;
    auto carve = [&](size_t bytes) -> void* {
        void* p = (void*)w;
        w += (bytes + 255) & ~(size_t)255;
        return p;
    };
    float*     deg = (float*)carve((size_t)kNT * 4);
    float*     vn  = (float*)carve((size_t)E * 4);
    float*     Xp  = (float*)carve((size_t)kNT * 4);
    float*     Z1  = (float*)carve((size_t)kNT * 4);
    float*     Z2  = (float*)carve((size_t)kNT * 4);
    float*     H1  = (float*)carve((size_t)kNT * kHID * 4);
    float*     S1  = (float*)carve((size_t)kNT * kHID * 4);
    float*     S2  = (float*)carve((size_t)kNT * kHID * 4);
    float*     H2  = (float*)carve((size_t)kNT * kHID * 4);
    _Float16*  Zc  = (_Float16*)carve((size_t)kNT * kD2 * 2);
    _Float16*  W2T = (_Float16*)carve((size_t)kD2 * kHID * 2);

    auto eg = [](int n) { return dim3((n + 255) / 256); };

    // ---- batch-invariant precompute: degrees, normalized edge values, W2^T ----
    hipMemsetAsync(deg, 0, (size_t)kNT * 4, stream);
    for (int g = 0; g < 4; ++g)
        deg_accum<<<eg(ne[g]), 256, 0, stream>>>(deg, rg[g], ne[g], sg[g]);
    for (int g = 0; g < 4; ++g)
        edge_norm<<<eg(ne[g]), 256, 0, stream>>>(vn + off[g], deg, rg[g], cg[g], ne[g], sg[g]);
    cvt_w2t<<<eg(kD2 * kHID), 256, 0, stream>>>(W2T, W2);

    const int nt64 = kNT * kHID;
    for (int b = 0; b < kB; ++b) {
        transpose_x<<<eg(kNT), 256, 0, stream>>>(Xp, x, b);

        // layer 1 powers (F=1)
        hipMemsetAsync(Z1, 0, (size_t)kNT * 4, stream);
        for (int g = 0; g < 4; ++g)
            spmm_f1<<<eg(ne[g]), 256, 0, stream>>>(Z1, Xp, vn + off[g], rg[g], cg[g], ne[g]);
        hipMemsetAsync(Z2, 0, (size_t)kNT * 4, stream);
        for (int g = 0; g < 4; ++g)
            spmm_f1<<<eg(ne[g]), 256, 0, stream>>>(Z2, Z1, vn + off[g], rg[g], cg[g], ne[g]);

        layer1<<<eg(nt64), 256, 0, stream>>>(H1, Xp, Z1, Z2, W1, b1);

        // layer 2 powers (F=64)
        hipMemsetAsync(S1, 0, (size_t)nt64 * 4, stream);
        for (int g = 0; g < 4; ++g) {
            long long tot = (long long)ne[g] * kHID;
            spmm_f64<<<dim3((unsigned)((tot + 255) / 256)), 256, 0, stream>>>(
                S1, H1, vn + off[g], rg[g], cg[g], ne[g]);
        }
        hipMemsetAsync(S2, 0, (size_t)nt64 * 4, stream);
        for (int g = 0; g < 4; ++g) {
            long long tot = (long long)ne[g] * kHID;
            spmm_f64<<<dim3((unsigned)((tot + 255) / 256)), 256, 0, stream>>>(
                S2, S1, vn + off[g], rg[g], cg[g], ne[g]);
        }

        // concat -> f16, WMMA GEMM + bias + relu
        cvt_cat<<<eg(nt64), 256, 0, stream>>>(Zc, H1, S1, S2);
        gemm_l2_wmma<<<dim3((kNT / 16 + 7) / 8), 256, 0, stream>>>(H2, Zc, W2T, b2);

        // pool + head
        head_pool<<<eg(kN), 256, 0, stream>>>((float*)d_out + (size_t)b * kN, H2, Wh, bh);
    }
}
